// SuperPixelClustering_6648609374642
// MI455X (gfx1250) — compile-verified, matrix operand order checked by reading, not run on hardware
//
#include <hip/hip_runtime.h>
#include <hip/hip_bf16.h>

typedef __attribute__((ext_vector_type(16))) _Float16 v16h;
typedef __attribute__((ext_vector_type(8)))  _Float16 v8h;
typedef __attribute__((ext_vector_type(8)))  float    v8f;
typedef __attribute__((ext_vector_type(4)))  float    vf4;

#define IMG_H 1024
#define IMG_W 1024
#define NIMG 8
#define NCFG 9
#define TOTSEG 815
#define NJOBS 85              // sum of gc over all 9 configs
#define SAL_ELEMS (8*3*1024*1024)

// grid configs for n = {20,30,40,65,80,95,125,150,175}
__constant__ int cGr[NCFG]     = {4, 5, 6, 8, 9, 10, 11, 12, 13};
__constant__ int cGc[NCFG]     = {5, 6, 7, 9, 9, 10, 12, 13, 14};
__constant__ int cSegOff[NCFG] = {0, 20, 50, 92, 164, 245, 345, 477, 633};

// ws layout (floats): [0,8) complexity sums, [8,16) tier ints, [16,16+8*815) seg sums->means
#define WS_TIER 8
#define WS_SEG  16

__device__ __forceinline__ unsigned lds_addr_of(const void* p) {
  return (unsigned)(unsigned long long)(size_t)p;
}

// -------- Kernel 1: stream image once; complexity + all-config segment sums --------
__global__ __launch_bounds__(256) void mainPassKernel(const float* __restrict__ img,
                                                      float* __restrict__ ws) {
  __shared__ float    stage[3 * 2 * IMG_W];                    // 24 KB: 2 rows x 3 channels
  __shared__ __align__(16) _Float16 chanMeanH[16 * IMG_W];     // 32 KB: 16 rows f16

  const int tid  = threadIdx.x;
  const int lane = tid & 31;
  const int wave = tid >> 5;
  const int bimg = blockIdx.x >> 6;              // 64 tiles per image
  const int tile = blockIdx.x & 63;
  const int y0   = tile << 4;                    // 16 rows per tile
  const float* base = img + (size_t)bimg * 3 * IMG_H * IMG_W;

  float acc = 0.0f;                              // per-thread sum of (dq)^2

  for (int ch = 0; ch < 8; ++ch) {               // 8 chunks of 2 rows
    const int yc = y0 + ch * 2;
    const unsigned ldsBase = lds_addr_of(&stage[0]);
    // async DMA: global -> LDS, 1536 x b128 (24 KB), GVS addressing
    for (int q = tid; q < 1536; q += 256) {
      int e4  = q << 2;                          // float index within chunk
      int cn  = e4 >> 11;                        // channel (2048 floats / channel)
      int rem = e4 & 2047;
      int rr  = rem >> 10;
      int x   = rem & 1023;
      unsigned goff  = (unsigned)((cn * (IMG_H * IMG_W) + (yc + rr) * IMG_W + x) * 4);
      unsigned laddr = ldsBase + (unsigned)(q << 4);
      asm volatile("global_load_async_to_lds_b128 %0, %1, %2"
                   :: "v"(laddr), "v"(goff), "s"(base) : "memory");
    }
    asm volatile("s_wait_asynccnt 0" ::: "memory");
    __syncthreads();

    // fused: channel-mean (for region means) + gray quantized contrast
#pragma unroll
    for (int i = 0; i < 8; ++i) {
      int p  = tid + (i << 8);
      int rr = p >> 10;
      int x  = p & 1023;
      float R  = stage[rr * 1024 + x];
      float G  = stage[2048 + rr * 1024 + x];
      float Bc = stage[4096 + rr * 1024 + x];
      chanMeanH[(ch * 2 + rr) * 1024 + x] = (_Float16)((R + G + Bc) * (1.0f / 3.0f));
      float g0 = 0.299f * R + 0.587f * G + 0.114f * Bc;
      int q0 = (int)(g0 * 255.0f); q0 = min(max(q0, 0), 255);
      if (x < 1023) {
        float R1 = stage[rr * 1024 + x + 1];
        float G1 = stage[2048 + rr * 1024 + x + 1];
        float B1 = stage[4096 + rr * 1024 + x + 1];
        float g1 = 0.299f * R1 + 0.587f * G1 + 0.114f * B1;
        int q1 = (int)(g1 * 255.0f); q1 = min(max(q1, 0), 255);
        int dq = q0 - q1;
        acc += (float)(dq * dq);
      }
    }
    __syncthreads();
  }

  // -------- Phase B: WMMA band sums: D = M_tile(16 x K) * ones(K x 16) --------
  v16h bones;
#pragma unroll
  for (int e = 0; e < 16; ++e) bones[e] = (_Float16)1.0f;
  const int m      = lane & 15;
  const int kup    = (lane >> 4) << 3;           // upper half-lanes handle K+8
  const int rowOff = m * 1024;

  for (int jb = wave; jb < NJOBS; jb += 8) {     // wave-uniform job loop
    int cfg = 0, cum = 0;
    while (jb >= cum + cGc[cfg]) { cum += cGc[cfg]; ++cfg; }
    int band = jb - cum;
    // force wave-uniform scalars so loop control is SALU, not EXEC masking
    band = __builtin_amdgcn_readfirstlane(band);
    const int gr      = __builtin_amdgcn_readfirstlane(cGr[cfg]);
    const int gc      = __builtin_amdgcn_readfirstlane(cGc[cfg]);
    const int segbase = __builtin_amdgcn_readfirstlane(cSegOff[cfg]);
    const int cstart  = __builtin_amdgcn_readfirstlane((band * 1024 + gc - 1) / gc);
    const int cend    = __builtin_amdgcn_readfirstlane(((band + 1) * 1024 + gc - 1) / gc);

    v8f c = {};
    for (int k0 = cstart & ~31; k0 < cend; k0 += 32) {
      // each lane's 16 A-elements = two contiguous aligned 8-half runs (2 x ds b128)
      v8h lo = *(const v8h*)&chanMeanH[rowOff + k0 + kup];
      v8h hi = *(const v8h*)&chanMeanH[rowOff + k0 + kup + 16];
      v16h a;
      if (k0 >= cstart && k0 + 32 <= cend) {     // uniform interior fast path: no mask
#pragma unroll
        for (int e = 0; e < 8; ++e) { a[e] = lo[e]; a[e + 8] = hi[e]; }
      } else {                                   // edge step: VALU selects, no branches
#pragma unroll
        for (int e = 0; e < 8; ++e) {
          int c0 = k0 + kup + e;
          int c1 = k0 + kup + 16 + e;
          a[e]     = (c0 >= cstart && c0 < cend) ? lo[e] : (_Float16)0.0f;
          a[e + 8] = (c1 >= cstart && c1 < cend) ? hi[e] : (_Float16)0.0f;
        }
      }
      c = __builtin_amdgcn_wmma_f32_16x16x32_f16(false, a, false, bones,
                                                 (short)0, c, false, false);
    }
    // D layout: VGPR i holds M=i (lanes 0-15) and M=i+8 (lanes 16-31); cols replicated
    if ((lane & 15) == 0) {
      int mb = (lane >> 4) << 3;
#pragma unroll
      for (int i = 0; i < 8; ++i) {
        int y  = y0 + mb + i;
        int ry = (y * gr) >> 10;
        int seg = segbase + ry * gc + band;
        atomicAdd(&ws[WS_SEG + bimg * TOTSEG + seg], c[i]);
      }
    }
  }

  // -------- Phase C: complexity reduction (wave32 shuffle + one atomic/wave) ----
#pragma unroll
  for (int off = 16; off > 0; off >>= 1) acc += __shfl_xor(acc, off, 32);
  if (lane == 0) atomicAdd(&ws[bimg], acc);
}

// -------- Kernel 2: tier select, cluster_sets, seg sums -> means --------
__global__ __launch_bounds__(256) void finalizeKernel(float* __restrict__ ws,
                                                      float* __restrict__ out) {
  const int bimg = blockIdx.x;
  const int tid  = threadIdx.x;
  if (tid == 0) {
    float comp = ws[bimg] / (1024.0f * 1023.0f);   // mean of (dq)^2 over pairs
    out[SAL_ELEMS + bimg] = comp;
    int tier = comp < 50.0f ? 0 : (comp < 150.0f ? 1 : 2);
    ((int*)ws)[WS_TIER + bimg] = tier;
    const int Bk[3] = {30, 80, 150};
    const int Dk[3] = {10, 15, 25};
    int k = Bk[tier], d = Dk[tier];
    int* ci = (int*)out + SAL_ELEMS + NIMG + bimg * 3;
    ci[0] = k - d; ci[1] = k; ci[2] = k + d;
  }
  for (int s = tid; s < TOTSEG; s += 256) {
    int cfg = 0;
    while (cfg + 1 < NCFG && s >= cSegOff[cfg + 1]) ++cfg;
    int local = s - cSegOff[cfg];
    int gr = cGr[cfg], gc = cGc[cfg];
    int r = local / gc, cc = local % gc;
    int cntR = ((r + 1) * 1024 + gr - 1) / gr - (r * 1024 + gr - 1) / gr;
    int cntC = ((cc + 1) * 1024 + gc - 1) / gc - (cc * 1024 + gc - 1) / gc;
    float* p = &ws[WS_SEG + bimg * TOTSEG + s];
    *p = *p / (float)(cntR * cntC);
  }
}

// -------- Kernel 3: saliency fill (write 96 MB, nontemporal) --------
__global__ __launch_bounds__(256) void fillKernel(const float* __restrict__ ws,
                                                  float* __restrict__ out) {
  const int bid  = blockIdx.x;
  const int bimg = bid / 3072;
  const int rem  = bid % 3072;
  const int j    = rem >> 10;       // which of the 3 cluster maps
  const int y    = rem & 1023;      // image row
  const int tier = ((const int*)ws)[WS_TIER + bimg];
  const int cfg  = tier * 3 + j;
  const int gr = cGr[cfg], gc = cGc[cfg];
  const int ry = (y * gr) >> 10;
  const float* means = ws + WS_SEG + bimg * TOTSEG + cSegOff[cfg] + ry * gc;

  __shared__ float sm[16];
  if ((int)threadIdx.x < gc) sm[threadIdx.x] = means[threadIdx.x];
  __syncthreads();

  const int x0 = threadIdx.x << 2;
  vf4 v;
#pragma unroll
  for (int i = 0; i < 4; ++i) {
    int x = x0 + i;
    v[i] = sm[(x * gc) >> 10];
  }
  float* dst = out + (((size_t)(bimg * 3 + j) * 1024 + y) * 1024 + x0);
  __builtin_nontemporal_store(v, (vf4*)dst);
}

extern "C" void kernel_launch(void* const* d_in, const int* in_sizes, int n_in,
                              void* d_out, int out_size, void* d_ws, size_t ws_size,
                              hipStream_t stream) {
  const float* img = (const float*)d_in[0];
  float* out = (float*)d_out;
  float* ws  = (float*)d_ws;
  (void)in_sizes; (void)n_in; (void)out_size; (void)ws_size;

  // zero accumulators (complexity sums + segment sums) each launch
  hipMemsetAsync(d_ws, 0, (size_t)(WS_SEG + NIMG * TOTSEG) * sizeof(float), stream);

  mainPassKernel<<<NIMG * 64, 256, 0, stream>>>(img, ws);
  finalizeKernel<<<NIMG, 256, 0, stream>>>(ws, out);
  fillKernel<<<NIMG * 3 * 1024, 256, 0, stream>>>(ws, out);
}
